// Temper_12601434047083
// MI455X (gfx1250) — compile-verified, matrix operand order checked by reading, not run on hardware
//
#include <hip/hip_runtime.h>
#include <math.h>

// ---------- types ----------
typedef __attribute__((ext_vector_type(16))) __bf16 bf16x16;
typedef __attribute__((ext_vector_type(8)))  __bf16 bf16x8;
typedef __attribute__((ext_vector_type(8)))  float  f32x8;

// Load one 16-wide bf16 WMMA fragment row-piece: two contiguous 16B chunks
// at p and p+16 elements (K-chunking for 16-bit A/B operands on gfx1250).
__device__ __forceinline__ bf16x16 load_frag(const __bf16* p) {
    bf16x8 lo = *(const bf16x8*)(p);
    bf16x8 hi = *(const bf16x8*)(p + 16);
    return __builtin_shufflevector(lo, hi, 0,1,2,3,4,5,6,7,8,9,10,11,12,13,14,15);
}

__device__ __forceinline__ void load_ab(bf16x16 (&a)[4], bf16x16 (&b)[2],
                                        const __bf16* const (&pa)[4],
                                        const __bf16* const (&pb)[2], int off) {
#pragma unroll
    for (int i = 0; i < 4; ++i) a[i] = load_frag(pa[i] + off);
#pragma unroll
    for (int j = 0; j < 2; ++j) b[j] = load_frag(pb[j] + off);
}

__device__ __forceinline__ void do_wmma(f32x8 (&acc)[4][2],
                                        const bf16x16 (&a)[4], const bf16x16 (&b)[2]) {
#pragma unroll
    for (int i = 0; i < 4; ++i)
#pragma unroll
        for (int j = 0; j < 2; ++j)
            acc[i][j] = __builtin_amdgcn_wmma_f32_16x16x32_bf16(
                false, a[i], false, b[j], (short)0, acc[i][j], false, false);
}

// ---------- WMMA GEMM: C[M,N] = act(A[M,K](bf16) @ W[N,K](bf16)^T + bias) ----------
// MODE 0: fused novelty epilogue: per-block partial of sum sqrt(max(0, xn+mn-2*acc))
// MODE 1: bias + relu, store bf16           -> intermediate activations
// MODE 2: bias + relu, store f32            -> final output
// Requires K % 64 == 0 and K >= 64 (all calls use K = 512).
template<int MODE>
__global__ __launch_bounds__(256)
void gemm_wmma(const __bf16* __restrict__ A, int lda,
               const __bf16* __restrict__ Wb, int ldw,
               const float* __restrict__ biasBase,
               const int* __restrict__ idxPtr, int idxSel,
               int wStride, int bStride,
               void* __restrict__ Cout, int ldc,
               int K,
               const float* __restrict__ xnorm,   // MODE 0 only
               const float* __restrict__ mnorm,   // MODE 0 only
               float* __restrict__ partials,      // MODE 0 only
               int validN)                        // MODE 0 only
{
    int e = 0;
    if (idxPtr) e = idxPtr[idxSel];                    // uniform scalar load
    const __bf16* Wp = Wb + (size_t)e * (size_t)wStride;
    const float*  bias = biasBase ? (biasBase + (size_t)e * (size_t)bStride) : nullptr;

    const int lane = threadIdx.x & 31;
    const int wave = threadIdx.x >> 5;                 // 0..7
    const int wm   = wave & 3;                         // 4 waves along M
    const int wn   = wave >> 2;                        // 2 waves along N
    const int r    = lane & 15;
    const int hi   = lane >> 4;

    const int waveRow = blockIdx.y * 256 + wm * 64;    // 4 x 16-row subtiles
    const int waveCol = blockIdx.x * 64  + wn * 32;    // 2 x 16-col subtiles

    // hoisted per-lane fragment base pointers (loop adds only the K offset)
    const __bf16* pa[4];
    const __bf16* pb[2];
#pragma unroll
    for (int i = 0; i < 4; ++i)
        pa[i] = A + (size_t)(waveRow + 16*i + r) * lda + hi*8;
#pragma unroll
    for (int j = 0; j < 2; ++j)
        pb[j] = Wp + (size_t)(waveCol + 16*j + r) * ldw + hi*8;

    f32x8 acc[4][2] = {};
    bf16x16 a0[4], b0[2], a1[4], b1[2];

    // software-pipelined K loop (double buffered, step 32, unrolled by 2)
    load_ab(a0, b0, pa, pb, 0);
    int k = 0;
    for (; k < K - 64; k += 64) {
        load_ab(a1, b1, pa, pb, k + 32);
        do_wmma(acc, a0, b0);
        load_ab(a0, b0, pa, pb, k + 64);
        do_wmma(acc, a1, b1);
    }
    load_ab(a1, b1, pa, pb, K - 32);
    do_wmma(acc, a0, b0);
    do_wmma(acc, a1, b1);

    __shared__ float sred[8];

    if (MODE == 0) {
        // fused novelty: sum over valid (row, col) of sqrt(||x||^2+||m||^2-2*G)
        float lsum = 0.f;
#pragma unroll
        for (int j = 0; j < 2; ++j) {
            const int col = waveCol + 16*j + r;
            if (col < validN) {
                const float mnc = mnorm[col];
#pragma unroll
                for (int i = 0; i < 4; ++i) {
#pragma unroll
                    for (int v = 0; v < 8; ++v) {
                        const int row = waveRow + 16*i + v + 8*hi;
                        float d2 = xnorm[row] + mnc - 2.f * acc[i][j][v];
                        lsum += sqrtf(fmaxf(d2, 0.f));
                    }
                }
            }
        }
        for (int o = 16; o > 0; o >>= 1) lsum += __shfl_down(lsum, o, 32);
        if (lane == 0) sred[wave] = lsum;
        __syncthreads();
        if (threadIdx.x < 32) {
            float s = (threadIdx.x < 8) ? sred[threadIdx.x] : 0.f;
            for (int o = 4; o > 0; o >>= 1) s += __shfl_down(s, o, 32);
            if (threadIdx.x == 0) partials[blockIdx.y] = s;
        }
        return;
    }

#pragma unroll
    for (int i = 0; i < 4; ++i) {
#pragma unroll
        for (int j = 0; j < 2; ++j) {
            const int col = waveCol + 16*j + r;
            const float bv = bias[col];
#pragma unroll
            for (int v = 0; v < 8; ++v) {
                const int row = waveRow + 16*i + v + 8*hi;
                float val = fmaxf(acc[i][j][v] + bv, 0.f);
                if (MODE == 1)
                    ((__bf16*)Cout)[(size_t)row * ldc + col] = (__bf16)val;
                else
                    ((float*)Cout)[(size_t)row * ldc + col] = val;
            }
        }
    }
}

// ---------- prep kernels ----------
__global__ void routing_kernel(const float* __restrict__ logits, int E, int* __restrict__ idx) {
    if (threadIdx.x == 0 && blockIdx.x == 0) {
        int a = 0;
        for (int i = 1; i < E; ++i) if (logits[i] > logits[a]) a = i;
        int b = -1;
        for (int i = 0; i < E; ++i) {
            if (i == a) continue;
            if (b < 0 || logits[i] > logits[b]) b = i;
        }
        idx[0] = a; idx[1] = b;
    }
}

// fp32 -> bf16 conversion of x, fused with per-row squared L2 norm (fp32)
__global__ void convert_x(const float* __restrict__ x, __bf16* __restrict__ xb,
                          float* __restrict__ xn, int D) {
    const int row  = blockIdx.x * 8 + (threadIdx.x >> 5);
    const int lane = threadIdx.x & 31;
    const float* src = x + (size_t)row * D;
    __bf16* dst = xb + (size_t)row * D;
    float s = 0.f;
    for (int i = lane; i < D; i += 32) {
        float v = src[i];
        s += v * v;
        dst[i] = (__bf16)v;
    }
    for (int o = 16; o > 0; o >>= 1) s += __shfl_down(s, o, 32);
    if (lane == 0) xn[row] = s;
}

// mem [Mr,D] -> bf16 padded to [Mp,D] (zero rows), plus squared norms
__global__ void convert_mem(const float* __restrict__ mem, __bf16* __restrict__ mb,
                            float* __restrict__ mn, int D, int Mr) {
    const int row  = blockIdx.x * 8 + (threadIdx.x >> 5);
    const int lane = threadIdx.x & 31;
    __bf16* dst = mb + (size_t)row * D;
    float s = 0.f;
    if (row < Mr) {
        const float* src = mem + (size_t)row * D;
        for (int i = lane; i < D; i += 32) {
            float v = src[i];
            s += v * v;
            dst[i] = (__bf16)v;
        }
    } else {
        for (int i = lane; i < D; i += 32) dst[i] = (__bf16)0.f;
    }
    for (int o = 16; o > 0; o >>= 1) s += __shfl_down(s, o, 32);
    if (lane == 0) mn[row] = s;
}

__global__ void convert_w(const float* __restrict__ pW, const float* __restrict__ oW,
                          __bf16* __restrict__ pWb, __bf16* __restrict__ oWb,
                          int nP, int nO) {
    const int stride = gridDim.x * blockDim.x;
    for (int k = blockIdx.x * blockDim.x + threadIdx.x; k < nP; k += stride)
        pWb[k] = (__bf16)pW[k];
    for (int k = blockIdx.x * blockDim.x + threadIdx.x; k < nO; k += stride)
        oWb[k] = (__bf16)oW[k];
}

// ---------- final novelty reduce (32 partials) ----------
__global__ void novelty_final(const float* __restrict__ partials, int n,
                              float* __restrict__ out_loc, float inv_count) {
    float s = (threadIdx.x < n) ? partials[threadIdx.x] : 0.f;
    __shared__ float sdata[8];
    for (int o = 16; o > 0; o >>= 1) s += __shfl_down(s, o, 32);
    if ((threadIdx.x & 31) == 0) sdata[threadIdx.x >> 5] = s;
    __syncthreads();
    if (threadIdx.x == 0) {
        float t = 0.f;
        for (int i = 0; i < 8; ++i) t += sdata[i];
        *out_loc = fminf(1.5f, t * inv_count);
    }
}

// ---------- launch ----------
extern "C" void kernel_launch(void* const* d_in, const int* in_sizes, int n_in,
                              void* d_out, int out_size, void* d_ws, size_t ws_size,
                              hipStream_t stream) {
    const float* x   = (const float*)d_in[0];   // [B, D]
    const float* mem = (const float*)d_in[1];   // [Mr, D]
    const float* rl  = (const float*)d_in[2];   // [E]
    const float* pW  = (const float*)d_in[3];   // [H, D]
    const float* pb  = (const float*)d_in[4];   // [H]
    const float* oW  = (const float*)d_in[5];   // [E, H, H]
    const float* ob  = (const float*)d_in[6];   // [E, H]

    const int B = 8192, D = 512, H = 512, Mr = 50, E = 3, Mp = 64;

    char* ws = (char*)d_ws;
    __bf16* xb   = (__bf16*)(ws + (size_t) 0 * 1024 * 1024);   // 8 MB
    __bf16* h0   = (__bf16*)(ws + (size_t) 8 * 1024 * 1024);   // 8 MB
    __bf16* h1   = (__bf16*)(ws + (size_t)16 * 1024 * 1024);   // 8 MB
    __bf16* pWb  = (__bf16*)(ws + (size_t)24 * 1024 * 1024);   // 0.5 MB
    __bf16* oWb  = (__bf16*)(ws + (size_t)25 * 1024 * 1024);   // 1.5 MB
    __bf16* memb = (__bf16*)(ws + (size_t)27 * 1024 * 1024);   // 64 KB
    float*  xn   = (float*) (ws + (size_t)28 * 1024 * 1024);   // 32 KB
    float*  mn   = (float*) (ws + (size_t)28 * 1024 * 1024 + 64 * 1024);
    float*  part = (float*) (ws + (size_t)28 * 1024 * 1024 + 128 * 1024);
    int*    idx  = (int*)   (ws + (size_t)28 * 1024 * 1024 + 256 * 1024);

    // 1) routing indices (top-2 of 3 logits; softmax is monotone)
    routing_kernel<<<1, 32, 0, stream>>>(rl, E, idx);

    // 2) precision conversion + norms
    convert_x  <<<B / 8, 256, 0, stream>>>(x, xb, xn, D);
    convert_w  <<<512,   256, 0, stream>>>(pW, oW, pWb, oWb, H * D, E * H * H);
    convert_mem<<<Mp / 8, 256, 0, stream>>>(mem, memb, mn, D, Mr);

    // 3) novelty: G = x @ mem^T fused with sqrt-sum epilogue -> 32 partials
    gemm_wmma<0><<<dim3(Mp / 64, B / 256), 256, 0, stream>>>(
        xb, D, memb, D, nullptr, nullptr, 0, 0, 0, nullptr, Mp, D,
        xn, mn, part, Mr);
    novelty_final<<<1, 256, 0, stream>>>(part, B / 256,
        (float*)d_out + (size_t)(out_size - 1), 1.f / ((float)B * (float)Mr));

    // 4) h0 = relu(x @ projW^T + proj_b)            [store bf16]
    gemm_wmma<1><<<dim3(H / 64, B / 256), 256, 0, stream>>>(
        xb, D, pWb, D, pb, nullptr, 0, 0, 0, h0, H, D,
        nullptr, nullptr, nullptr, 0);

    // 5) h1 = relu(h0 @ W[a]^T + b[a])              [store bf16]
    gemm_wmma<1><<<dim3(H / 64, B / 256), 256, 0, stream>>>(
        h0, H, oWb, H, ob, idx, 0, H * H, H, h1, H, H,
        nullptr, nullptr, nullptr, 0);

    // 6) out = relu(h1 @ W[b]^T + b[b])             [store f32 -> d_out]
    gemm_wmma<2><<<dim3(H / 64, B / 256), 256, 0, stream>>>(
        h1, H, oWb, H, ob, idx, 1, H * H, H, (float*)d_out, H, H,
        nullptr, nullptr, nullptr, 0);
}
